// MLA_52115133170132
// MI455X (gfx1250) — matrix operand order repaired and checked by reading, not compiled, not run on hardware
//
#include <hip/hip_runtime.h>
#include <hip/hip_bf16.h>
#include <cstdint>

typedef __bf16 bf16_t;
typedef __attribute__((ext_vector_type(16))) __bf16 v16bf;
typedef __attribute__((ext_vector_type(8)))  __bf16 v8bf;
typedef __attribute__((ext_vector_type(8)))  float  v8f;

union BF16x16 { v16bf v; v8bf h[2]; };

__device__ __forceinline__ v8f wmma_bf16(v16bf a, v16bf b, v8f c) {
    // D = A(16x32 bf16) * B(32x16 bf16) + C(16x16 f32)
    return __builtin_amdgcn_wmma_f32_16x16x32_bf16(
        false, a, false, b, (short)0, c, false, false);
}

__device__ __forceinline__ v16bf ld_frag(const bf16_t* p0, const bf16_t* p1) {
    BF16x16 u;
    u.h[0] = *(const v8bf*)p0;
    u.h[1] = *(const v8bf*)p1;
    return u.v;
}

// Async 16B copy global -> LDS (ASYNCcnt-tracked, in-order completion).
__device__ __forceinline__ void async_b128(uint32_t lds_byte, const void* sbase, int voff) {
    asm volatile("global_load_async_to_lds_b128 %0, %1, %2"
                 :: "v"(lds_byte), "v"(voff), "s"(sbase) : "memory");
}
__device__ __forceinline__ uint32_t lds_addr(const void* p) {
    return (uint32_t)(uintptr_t)p;   // generic addr low 32 bits == LDS byte offset
}

// ---------------------------------------------------------------------------
// GEMM: C[M,N] = A[M,K] @ B[K,N]; A bf16 MxK, BT bf16 NxK (pre-transposed).
// Block = 256 threads = 8 waves (4x2); block tile BM x 128, wave tile
// (BM/4) x 64. Double-buffered async-LDS pipeline. M%BM==0, K%32==0.
// ---------------------------------------------------------------------------
template <int BM, typename OutT>
__global__ __launch_bounds__(256) void gemm_async_kernel(
    const bf16_t* __restrict__ A, const bf16_t* __restrict__ BT,
    OutT* __restrict__ C, int M, int N, int K)
{
    constexpr int MI = BM / 64;                 // 16-row A frags per wave
    __shared__ bf16_t As[2][BM * 40];
    __shared__ bf16_t Bs[2][128 * 40];

    const int t    = threadIdx.x;
    const int wid  = t >> 5, lane = t & 31;
    const int wm   = wid & 3, wn = wid >> 2;
    const int half = lane >> 4, l16 = lane & 15;
    const int m0   = blockIdx.x * BM, n0 = blockIdx.y * 128;

    const int pitch = K * 2;                    // row pitch bytes (A and BT)
    int lrowA, lcolA;
    if constexpr (BM == 256) { lrowA = t;      lcolA = 0; }
    else                     { lrowA = t >> 1; lcolA = (t & 1) * 32; }
    const int lrowB = t >> 1, lcolB = (t & 1) * 32;

    const int gAbase = (m0 + lrowA) * pitch + lcolA;
    const int gBbase = (n0 + lrowB) * pitch + lcolB;

    v8f acc[MI][4] = {};
    const int nk = K >> 5;

    auto issue = [&](int buf, int k0) {
        const int ka = k0 * 2;
        const uint32_t lA = lds_addr(&As[buf][lrowA * 40]) + lcolA;
        const uint32_t lB = lds_addr(&Bs[buf][lrowB * 40]) + lcolB;
        async_b128(lA,      A,  gAbase + ka);
        async_b128(lA + 16, A,  gAbase + ka + 16);
        if constexpr (BM == 256) {
            async_b128(lA + 32, A, gAbase + ka + 32);
            async_b128(lA + 48, A, gAbase + ka + 48);
        }
        async_b128(lB,      BT, gBbase + ka);
        async_b128(lB + 16, BT, gBbase + ka + 16);
    };

    issue(0, 0);
    for (int kk = 0; kk < nk; ++kk) {
        if (kk + 1 < nk) {
            issue((kk + 1) & 1, (kk + 1) << 5);
            if constexpr (BM == 256)
                asm volatile("s_wait_asynccnt 0x6" ::: "memory");  // tile kk landed
            else
                asm volatile("s_wait_asynccnt 0x4" ::: "memory");
        } else {
            asm volatile("s_wait_asynccnt 0x0" ::: "memory");
        }
        __syncthreads();

        const bf16_t* as = As[kk & 1];
        const bf16_t* bs = Bs[kk & 1];
        v16bf af[MI], bfarr[4];
        #pragma unroll
        for (int i = 0; i < MI; ++i) {
            const bf16_t* base = &as[(wm * (MI * 16) + i * 16 + l16) * 40 + half * 8];
            af[i] = ld_frag(base, base + 16);
        }
        #pragma unroll
        for (int j = 0; j < 4; ++j) {
            const bf16_t* base = &bs[(wn * 64 + j * 16 + l16) * 40 + half * 8];
            bfarr[j] = ld_frag(base, base + 16);
        }
        #pragma unroll
        for (int i = 0; i < MI; ++i)
            #pragma unroll
            for (int j = 0; j < 4; ++j)
                acc[i][j] = wmma_bf16(af[i], bfarr[j], acc[i][j]);
        __syncthreads();                        // readers done -> buffer reusable
    }

    #pragma unroll
    for (int i = 0; i < MI; ++i)
        #pragma unroll
        for (int j = 0; j < 4; ++j) {
            const int col = n0 + wn * 64 + j * 16 + l16;
            if (col < N) {
                #pragma unroll
                for (int r = 0; r < 8; ++r) {
                    const int row = m0 + wm * (MI * 16) + i * 16 + r + 8 * half;
                    C[(size_t)row * N + col] = (OutT)acc[i][j][r];
                }
            }
        }
}

// ---------------------------------------------------------------------------
__global__ __launch_bounds__(256) void convert_bf16_kernel(
    const float* __restrict__ in, bf16_t* __restrict__ out)
{
    const size_t i = ((size_t)blockIdx.x * 256 + threadIdx.x) * 4;
    const float4 f = *(const float4*)(in + i);
    out[i]     = (bf16_t)f.x;
    out[i + 1] = (bf16_t)f.y;
    out[i + 2] = (bf16_t)f.z;
    out[i + 3] = (bf16_t)f.w;
}

// Tiled transpose: out[c][r] = (bf16)in[r][c].  R,C % 32 == 0.
template <typename InT>
__global__ __launch_bounds__(256) void transpose_bf16_kernel(
    const InT* __restrict__ in, bf16_t* __restrict__ out, int R, int Ccols)
{
    __shared__ bf16_t tile[32][33];
    const int c0 = blockIdx.x * 32, r0 = blockIdx.y * 32;
    const int tx = threadIdx.x & 31, ty = threadIdx.x >> 5;
    #pragma unroll
    for (int i = 0; i < 4; ++i)
        tile[ty + i * 8][tx] = (bf16_t)in[(size_t)(r0 + ty + i * 8) * Ccols + c0 + tx];
    __syncthreads();
    #pragma unroll
    for (int i = 0; i < 4; ++i)
        out[(size_t)(c0 + ty + i * 8) * R + r0 + tx] = tile[tx][ty + i * 8];
}

// RMSNorm rows of 512: fp32 in, bf16 out. One block per row.
__global__ __launch_bounds__(256) void rmsnorm512_kernel(
    const float* __restrict__ x, const float* __restrict__ g, bf16_t* __restrict__ o)
{
    __shared__ float red[8];
    const float* p = x + (size_t)blockIdx.x * 512;
    bf16_t* q = o + (size_t)blockIdx.x * 512;
    const int t = threadIdx.x;
    const float v0 = p[t], v1 = p[t + 256];
    float ss = v0 * v0 + v1 * v1;
    #pragma unroll
    for (int off = 16; off; off >>= 1) ss += __shfl_xor(ss, off, 32);
    if ((t & 31) == 0) red[t >> 5] = ss;
    __syncthreads();
    float tot = 0.0f;
    #pragma unroll
    for (int i = 0; i < 8; ++i) tot += red[i];
    const float inv = rsqrtf(tot * (1.0f / 512.0f) + 1e-6f);
    q[t]       = (bf16_t)(v0 * inv * g[t]);
    q[t + 256] = (bf16_t)(v1 * inv * g[t + 256]);
}

// RoPE. ln(10000)/32 = 0.28782313662425572
__global__ void rope_k_kernel(const float* __restrict__ in, bf16_t* __restrict__ out)
{
    const int row = blockIdx.x;
    const int s = row & 2047;
    const int j = threadIdx.x;                        // 0..31
    const float f = __expf(-0.28782313662425572f * (float)j);
    float c, sn;  sincosf((float)s * f, &sn, &c);
    const float* p = in + (size_t)row * 64 + 2 * j;
    bf16_t* q = out + (size_t)row * 64 + 2 * j;
    const float x0 = p[0], x1 = p[1];
    q[0] = (bf16_t)(x0 * c - x1 * sn);
    q[1] = (bf16_t)(x1 * c + x0 * sn);
}

__global__ __launch_bounds__(512) void rope_q_kernel(float* __restrict__ q) // fp32 in place
{
    const int row = blockIdx.x;
    const int s = row & 2047;
    const int t = threadIdx.x;                        // 0..511
    const int h = t >> 5, j = t & 31;
    const float f = __expf(-0.28782313662425572f * (float)j);
    float c, sn;  sincosf((float)s * f, &sn, &c);
    float* p = q + (size_t)row * 3072 + h * 192 + 128 + 2 * j;
    const float x0 = p[0], x1 = p[1];
    p[0] = x0 * c - x1 * sn;
    p[1] = x1 * c + x0 * sn;
}

// ---------------------------------------------------------------------------
// Flash attention. Grid (S/128, H, B), block 128 = 4 waves; 32 q-rows/wave
// (2 m-frags). bf16 K/Kr/Vt streamed by double-buffered async-LDS loads.
// Causal mask only fires on the diagonal tile (kt == q0) and is a fixed
// lane pattern there: key_local > row_local, i.e. d > mi*16 + r (+/- 16).
// ---------------------------------------------------------------------------
__global__ __launch_bounds__(128) void attn_kernel(
    const float*  __restrict__ Q,    // (B*S, 3072) fp32, rope applied
    const bf16_t* __restrict__ Kn,   // (B*S, 2048) bf16, col = h*128+d
    const bf16_t* __restrict__ Kr,   // (B*S, 64)   bf16
    const bf16_t* __restrict__ Vt,   // (B,H,128,S) bf16 (pre-transposed)
    bf16_t* __restrict__ O)          // (B*S, 2048) bf16, col = h*128+d
{
    const int S = 2048;
    const int qb = blockIdx.x, h = blockIdx.y, b = blockIdx.z;
    const int t = threadIdx.x, wid = t >> 5, lane = t & 31;
    const int half = lane >> 4, l16 = lane & 15;
    const int dlane = l16 - 8 * half;           // diagonal-mask helper, [-8,15]

    __shared__ bf16_t sKn[2][32 * 136];
    __shared__ bf16_t sKr[2][32 * 72];
    __shared__ bf16_t sVt[2][128 * 40];
    __shared__ bf16_t sP[4][32 * 40];

    const float SCALE = 0.07216878364870323f;   // 1/sqrt(192)

    // Q fragments (scale folded in): 2 m-frags x 6 K-chunks covering d=0..191
    const int q0 = qb * 128 + wid * 32;
    v16bf qf[2][6];
    #pragma unroll
    for (int mi = 0; mi < 2; ++mi) {
        const float* qp = Q + (size_t)(b * S + q0 + mi * 16 + l16) * 3072 + h * 192;
        #pragma unroll
        for (int c = 0; c < 6; ++c) {
            const int d0 = 32 * c;
            BF16x16 u;
            #pragma unroll
            for (int i = 0; i < 8; ++i) {
                u.v[i]     = (bf16_t)(qp[d0 + 8 * half + i]      * SCALE);
                u.v[i + 8] = (bf16_t)(qp[d0 + 16 + 8 * half + i] * SCALE);
            }
            qf[mi][c] = u.v;
        }
    }

    v8f o[2][8] = {};
    float mrow[2][8], lrow[2][8];
    #pragma unroll
    for (int mi = 0; mi < 2; ++mi)
        #pragma unroll
        for (int r = 0; r < 8; ++r) { mrow[mi][r] = -3e38f; lrow[mi][r] = 0.0f; }

    const int key4 = t >> 2, part = t & 3;
    const int nt = (qb * 128 + 128) >> 5;

    auto issue = [&](int buf, int kt) {
        {   // Kn tile: 32 keys x 256B
            const int g = (b * S + kt + key4) * 4096 + h * 256 + part * 64;
            const uint32_t l = lds_addr(&sKn[buf][key4 * 136]) + part * 64;
            async_b128(l,      Kn, g);
            async_b128(l + 16, Kn, g + 16);
            async_b128(l + 32, Kn, g + 32);
            async_b128(l + 48, Kn, g + 48);
        }
        {   // Kr tile: 32 keys x 128B
            const int g = (b * S + kt + key4) * 128 + part * 32;
            const uint32_t l = lds_addr(&sKr[buf][key4 * 72]) + part * 32;
            async_b128(l,      Kr, g);
            async_b128(l + 16, Kr, g + 16);
        }
        {   // Vt tile: 128 d-rows x 64B (keys kt..kt+31)
            const int g = ((b * 16 + h) * 128 + t) * 4096 + kt * 2;
            const uint32_t l = lds_addr(&sVt[buf][t * 40]);
            async_b128(l,      Vt, g);
            async_b128(l + 16, Vt, g + 16);
            async_b128(l + 32, Vt, g + 32);
            async_b128(l + 48, Vt, g + 48);
        }
    };

    issue(0, 0);
    for (int it = 0; it < nt; ++it) {
        const int kt = it << 5;
        if (it + 1 < nt) {
            issue((it + 1) & 1, kt + 32);
            asm volatile("s_wait_asynccnt 0xa" ::: "memory");   // tile `it` landed
        } else {
            asm volatile("s_wait_asynccnt 0x0" ::: "memory");
        }
        __syncthreads();

        const bf16_t* kn = sKn[it & 1];
        const bf16_t* kr = sKr[it & 1];
        const bf16_t* vt = sVt[it & 1];
        const bool active = (kt <= q0 + 31);            // wave-uniform
        const bool diag   = (kt == q0);                 // only masked tile
        if (active) {
            v8f sc[2][2] = {};
            #pragma unroll
            for (int j = 0; j < 2; ++j) {
                const int key = j * 16 + l16;
                #pragma unroll
                for (int c = 0; c < 4; ++c) {
                    const bf16_t* base = &kn[key * 136 + c * 32 + half * 8];
                    const v16bf bfr = ld_frag(base, base + 16);
                    sc[0][j] = wmma_bf16(qf[0][c], bfr, sc[0][j]);
                    sc[1][j] = wmma_bf16(qf[1][c], bfr, sc[1][j]);
                }
                #pragma unroll
                for (int c = 0; c < 2; ++c) {
                    const bf16_t* base = &kr[key * 72 + c * 32 + half * 8];
                    const v16bf bfr = ld_frag(base, base + 16);
                    sc[0][j] = wmma_bf16(qf[0][4 + c], bfr, sc[0][j]);
                    sc[1][j] = wmma_bf16(qf[1][4 + c], bfr, sc[1][j]);
                }
            }
            #pragma unroll
            for (int mi = 0; mi < 2; ++mi) {
                #pragma unroll
                for (int r = 0; r < 8; ++r) {
                    float s0 = sc[mi][0][r], s1 = sc[mi][1][r];
                    if (diag) {
                        // key_local > row_local  (position-independent pattern)
                        if (dlane > mi * 16 + r)      s0 = -3e38f;
                        if (dlane > mi * 16 + r - 16) s1 = -3e38f;
                    }
                    float mt = fmaxf(s0, s1);
                    #pragma unroll
                    for (int off = 8; off; off >>= 1)
                        mt = fmaxf(mt, __shfl_xor(mt, off, 32));
                    const float mn = fmaxf(mrow[mi][r], mt);
                    const float p0 = __expf(s0 - mn), p1 = __expf(s1 - mn);
                    float rs = p0 + p1;
                    #pragma unroll
                    for (int off = 8; off; off >>= 1)
                        rs += __shfl_xor(rs, off, 32);
                    const float alpha = __expf(mrow[mi][r] - mn);
                    lrow[mi][r] = lrow[mi][r] * alpha + rs;
                    mrow[mi][r] = mn;
                    #pragma unroll
                    for (int nf = 0; nf < 8; ++nf) o[mi][nf][r] *= alpha;
                    sP[wid][(mi * 16 + r + 8 * half) * 40 + l16]      = (bf16_t)p0;
                    sP[wid][(mi * 16 + r + 8 * half) * 40 + 16 + l16] = (bf16_t)p1;
                }
            }
        }
        __syncthreads();
        if (active) {
            v16bf pf[2];
            #pragma unroll
            for (int mi = 0; mi < 2; ++mi) {
                const bf16_t* pb = &sP[wid][(mi * 16 + l16) * 40 + 8 * half];
                pf[mi] = ld_frag(pb, pb + 16);
            }
            #pragma unroll
            for (int nf = 0; nf < 8; ++nf) {
                const bf16_t* vb = &vt[(nf * 16 + l16) * 40 + half * 8];
                const v16bf bfr = ld_frag(vb, vb + 16);
                o[0][nf] = wmma_bf16(pf[0], bfr, o[0][nf]);
                o[1][nf] = wmma_bf16(pf[1], bfr, o[1][nf]);
            }
        }
        __syncthreads();                                // buffer reusable
    }

    #pragma unroll
    for (int mi = 0; mi < 2; ++mi)
        #pragma unroll
        for (int r = 0; r < 8; ++r) {
            const float inv = 1.0f / lrow[mi][r];
            bf16_t* dst = O + (size_t)(b * S + q0 + mi * 16 + r + 8 * half) * 2048 + h * 128;
            #pragma unroll
            for (int nf = 0; nf < 8; ++nf)
                dst[nf * 16 + l16] = (bf16_t)(o[mi][nf][r] * inv);
        }
}

// ---------------------------------------------------------------------------
extern "C" void kernel_launch(void* const* d_in, const int* in_sizes, int n_in,
                              void* d_out, int out_size, void* d_ws, size_t ws_size,
                              hipStream_t stream)
{
    (void)in_sizes; (void)n_in; (void)out_size; (void)ws_size;
    const float* x        = (const float*)d_in[0];
    const float* w_dq     = (const float*)d_in[1];
    const float* g_q      = (const float*)d_in[2];
    const float* w_uq     = (const float*)d_in[3];
    const float* w_dkv    = (const float*)d_in[4];
    const float* g_kv     = (const float*)d_in[5];
    const float* w_ukrope = (const float*)d_in[6];
    const float* w_uv     = (const float*)d_in[7];
    const float* w_uknope = (const float*)d_in[8];
    const float* w_o      = (const float*)d_in[9];
    float* out = (float*)d_out;

    char* wsp = (char*)d_ws;
    auto alloc = [&](size_t bytes) {
        void* p = wsp; wsp += (bytes + 255) & ~(size_t)255; return p;
    };
    bf16_t* x_bf      = (bf16_t*)alloc((size_t)4096 * 2048 * 2);
    bf16_t* wT_dkv    = (bf16_t*)alloc((size_t)512  * 2048 * 2);
    bf16_t* wT_dq     = (bf16_t*)alloc((size_t)512  * 2048 * 2);
    bf16_t* wT_uq     = (bf16_t*)alloc((size_t)3072 * 512  * 2);
    bf16_t* wT_ukrope = (bf16_t*)alloc((size_t)128  * 2048 * 2);  // 64 valid rows, padded
    bf16_t* wT_uknope = (bf16_t*)alloc((size_t)2048 * 512  * 2);
    bf16_t* wT_uv     = (bf16_t*)alloc((size_t)2048 * 512  * 2);
    bf16_t* wT_o      = (bf16_t*)alloc((size_t)2048 * 2048 * 2);
    float*  kvraw     = (float*) alloc((size_t)4096 * 512  * 4);
    float*  qlraw     = (float*) alloc((size_t)4096 * 512  * 4);
    float*  krraw     = (float*) alloc((size_t)4096 * 64   * 4);
    bf16_t* kv_bf     = (bf16_t*)alloc((size_t)4096 * 512  * 2);
    bf16_t* ql_bf     = (bf16_t*)alloc((size_t)4096 * 512  * 2);
    bf16_t* kr_bf     = (bf16_t*)alloc((size_t)4096 * 64   * 2);
    bf16_t* knope     = (bf16_t*)alloc((size_t)4096 * 2048 * 2);
    bf16_t* v_bf      = (bf16_t*)alloc((size_t)4096 * 2048 * 2);
    bf16_t* vT        = (bf16_t*)alloc((size_t)4096 * 2048 * 2);
    float*  qall      = (float*) alloc((size_t)4096 * 3072 * 4);
    bf16_t* aout      = (bf16_t*)alloc((size_t)4096 * 2048 * 2);

    // ---- one-time bf16 prep ----
    convert_bf16_kernel<<<8192, 256, 0, stream>>>(x, x_bf);
    auto tr = [&](const float* in, bf16_t* o2, int R, int C) {
        transpose_bf16_kernel<float><<<dim3(C / 32, R / 32), 256, 0, stream>>>(in, o2, R, C);
    };
    tr(w_dkv,    wT_dkv,    2048, 512);
    tr(w_dq,     wT_dq,     2048, 512);
    tr(w_uq,     wT_uq,     512,  3072);
    tr(w_ukrope, wT_ukrope, 2048, 64);
    tr(w_uknope, wT_uknope, 512,  2048);
    tr(w_uv,     wT_uv,     512,  2048);
    tr(w_o,      wT_o,      2048, 2048);

    auto gemm128F = [&](const bf16_t* A, const bf16_t* BT, float* C, int M, int N, int K) {
        gemm_async_kernel<128, float><<<dim3(M / 128, (N + 127) / 128), 256, 0, stream>>>(A, BT, C, M, N, K);
    };
    auto gemm256F = [&](const bf16_t* A, const bf16_t* BT, float* C, int M, int N, int K) {
        gemm_async_kernel<256, float><<<dim3(M / 256, N / 128), 256, 0, stream>>>(A, BT, C, M, N, K);
    };
    auto gemm256H = [&](const bf16_t* A, const bf16_t* BT, bf16_t* C, int M, int N, int K) {
        gemm_async_kernel<256, bf16_t><<<dim3(M / 256, N / 128), 256, 0, stream>>>(A, BT, C, M, N, K);
    };

    // ---- MLA pipeline ----
    gemm128F(x_bf, wT_dkv, kvraw, 4096, 512, 2048);
    rmsnorm512_kernel<<<4096, 256, 0, stream>>>(kvraw, g_kv, kv_bf);
    gemm128F(x_bf, wT_dq, qlraw, 4096, 512, 2048);
    rmsnorm512_kernel<<<4096, 256, 0, stream>>>(qlraw, g_q, ql_bf);
    gemm128F(x_bf, wT_ukrope, krraw, 4096, 64, 2048);
    rope_k_kernel<<<4096, 32, 0, stream>>>(krraw, kr_bf);
    gemm256H(kv_bf, wT_uknope, knope, 4096, 2048, 512);
    gemm256H(kv_bf, wT_uv, v_bf, 4096, 2048, 512);
    transpose_bf16_kernel<bf16_t><<<dim3(64, 64), 256, 0, stream>>>(v_bf, vT, 2048, 2048);
    transpose_bf16_kernel<bf16_t><<<dim3(64, 64), 256, 0, stream>>>(
        v_bf + (size_t)2048 * 2048, vT + (size_t)2048 * 2048, 2048, 2048);
    gemm256F(ql_bf, wT_uq, qall, 4096, 3072, 512);
    rope_q_kernel<<<4096, 512, 0, stream>>>(qall);
    attn_kernel<<<dim3(16, 16, 2), dim3(128), 0, stream>>>(qall, knope, kr_bf, vT, aout);
    gemm256F(aout, wT_o, out, 4096, 2048, 2048);
}
